// MambaEncoderLayer_19404662243338
// MI455X (gfx1250) — matrix operand order repaired and checked by reading, not compiled, not run on hardware
//
#include <hip/hip_runtime.h>
#include <hip/hip_bf16.h>

// ---------------------------------------------------------------------------
// MambaEncoderLayer for MI455X (gfx1250, wave32, WMMA 16x16x32 f16,
// async global->LDS double-buffered B-tiles)
// ---------------------------------------------------------------------------
#define BATCH    8
#define SEQ      2048
#define ROWS     (BATCH * SEQ)        // 16384
#define DMODEL   512
#define DINNER   1024
#define DSTATE   64
#define NHEADS   16
#define HEADDIM  64
#define DFF      2048
#define CONVDIM  1152                 // D_INNER + 2*D_STATE
#define DINPROJ  2192                 // 2*D_INNER + 2*D_STATE + NHEADS
#define NPAD1    2240                 // DINPROJ rounded up to 64
#define EPSF     1e-6f

typedef __attribute__((ext_vector_type(16))) _Float16 v16h;
typedef __attribute__((ext_vector_type(8)))  float    v8f;

// ---- d_out layout (floats): x | m | dt | z  (reference return order) ------
#define X_OFF   0
#define M_OFF   (ROWS * DMODEL)                  //  8,388,608
#define DT_OFF  (M_OFF + ROWS * DMODEL)          // 16,777,216
#define Z_OFF   (DT_OFF + ROWS * NHEADS)         // 17,039,360

// ---- workspace layout (bytes) ---------------------------------------------
#define WS_WTIN   ((size_t)0)                                   // f16 [2240][512]
#define WS_WTOUT  (WS_WTIN  + (size_t)NPAD1 * DMODEL * 2)       // f16 [512][1024]
#define WS_WT1    (WS_WTOUT + (size_t)DMODEL * DINNER * 2)      // f16 [2048][512]
#define WS_WT2    (WS_WT1   + (size_t)DFF * DMODEL * 2)         // f16 [512][2048]
#define WS_U16    (WS_WT2   + (size_t)DMODEL * DFF * 2)         // f16 [16384][512] (reused as h16)
#define WS_XRAW   (WS_U16   + (size_t)ROWS * DMODEL * 2)        // f32 [16384][1152] (reused as g16 f16)
#define WS_XCONV  (WS_XRAW  + (size_t)ROWS * CONVDIM * 4)       // f32 [16384][1152]
#define WS_DTRAW  (WS_XCONV + (size_t)ROWS * CONVDIM * 4)       // f32 [16384][16]
#define WS_Y      (WS_DTRAW + (size_t)ROWS * NHEADS * 4)        // f32 [16384][1024]
#define WS_Y16    (WS_Y     + (size_t)ROWS * DINNER * 4)        // f16 [16384][1024]

// LDS B-tile row stride in halves: 32 K-halves padded to 40 (80 B = 20 banks)
// -> 16-lane b128 reads land on disjoint 4-bank groups (optimal 2-cycle).
#define BSTRIDE  40

// ===========================================================================
// Weight transpose + f16 convert:  dst[n*K+k] = (f16) src[k*N+n], pad-zero n>=N
// ===========================================================================
__global__ void transpose_to_f16(const float* __restrict__ src, _Float16* __restrict__ dst,
                                 int K, int N, int Npad) {
    int idx = blockIdx.x * blockDim.x + threadIdx.x;
    int total = Npad * K;
    if (idx >= total) return;
    int n = idx / K, k = idx - n * K;
    dst[idx] = (n < N) ? (_Float16)src[(size_t)k * N + n] : (_Float16)0.0f;
}

// ===========================================================================
// Row-wise rmsnorm (optional mask, scale) -> f16
// ===========================================================================
__global__ void rmsnorm_f16_kernel(const float* __restrict__ src, const float* __restrict__ mask,
                                   float scale, int width, _Float16* __restrict__ out) {
    int row = blockIdx.x, tid = threadIdx.x;
    const float* s = src + (size_t)row * width;
    float m = mask ? mask[row] : 1.0f;
    float ss = 0.0f;
    for (int c = tid; c < width; c += 256) { float v = s[c] * scale * m; ss += v * v; }
    __shared__ float red[256];
    red[tid] = ss; __syncthreads();
    for (int st = 128; st > 0; st >>= 1) { if (tid < st) red[tid] += red[tid + st]; __syncthreads(); }
    float rs = rsqrtf(red[0] / (float)width + EPSF);
    for (int c = tid; c < width; c += 256) {
        float v = s[c] * scale * m;
        out[(size_t)row * width + c] = (_Float16)(v * rs);
    }
}

// ===========================================================================
// Async copy of one 16-byte chunk per lane: global -> LDS (ASYNCcnt-tracked)
// ===========================================================================
__device__ __forceinline__ void async_cp16(unsigned lds_addr, const _Float16* g) {
    asm volatile("global_load_async_to_lds_b128 %0, %1, off"
                 :: "v"(lds_addr), "v"(g) : "memory");
}

__device__ __forceinline__ void wait_async0() {
    asm volatile("s_wait_asynccnt 0x0" ::: "memory");
}

// Stage a 64(N) x 32(K) f16 B-tile into LDS (padded rows). 128 threads,
// 2 x b128 async copies each.
__device__ __forceinline__ void stage_B_async(const _Float16* __restrict__ Bw, int K,
                                              int n0, int k0, _Float16* sbuf) {
    const int t = threadIdx.x;            // 0..127
#pragma unroll
    for (int r = 0; r < 2; ++r) {
        int c  = t + r * 128;             // chunk id 0..255 (16B chunks)
        int n  = c >> 2;                  // row within tile
        int kc = c & 3;                   // 16B chunk within 64B row
        const _Float16* g = Bw + (size_t)(n0 + n) * K + k0 + kc * 8;
        unsigned lds = (unsigned)(uintptr_t)(sbuf + n * BSTRIDE + kc * 8);
        async_cp16(lds, g);
    }
}

// ===========================================================================
// Block-cooperative WMMA mainloop. 4 waves x 16(M) rows; shared 64(N) tile.
// B is double-buffered through LDS via async copies; A streamed from global.
// A frag (16-bit A 16x32): lanes 0-15 -> K[0..15];
//                          lanes 16-31 -> K[16..31].
// B frag (32x16):          lanes 0-15 -> K[0..15]; lanes 16-31 -> K[16..31].
// ===========================================================================
__device__ __forceinline__ void wmma_strip(const _Float16* __restrict__ A,
                                           const _Float16* __restrict__ Bw,
                                           int K, int m0, int n0,
                                           _Float16* sb0, _Float16* sb1, v8f acc[4]) {
    const int lane = threadIdx.x & 31;
    const int half = lane >> 4;
    const int lm   = lane & 15;
    const _Float16* arow = A + (size_t)(m0 + lm) * K + half * 16;

    stage_B_async(Bw, K, n0, 0, sb0);
    for (int k0 = 0; k0 < K; k0 += 32) {
        _Float16* cur = (k0 & 32) ? sb1 : sb0;
        _Float16* nxt = (k0 & 32) ? sb0 : sb1;
        wait_async0();
        __syncthreads();                       // all waves' async data visible
        if (k0 + 32 < K) stage_B_async(Bw, K, n0, k0 + 32, nxt);

        union { v16h h; uint4 u[2]; } ua;
        ua.u[0] = *(const uint4*)(arow + k0);
        ua.u[1] = *(const uint4*)(arow + k0 + 8);
#pragma unroll
        for (int j = 0; j < 4; ++j) {
            union { v16h h; uint4 u[2]; } ub;
            const _Float16* bl = cur + (j * 16 + lm) * BSTRIDE + half * 16;
            ub.u[0] = *(const uint4*)bl;
            ub.u[1] = *(const uint4*)(bl + 8);
            acc[j] = __builtin_amdgcn_wmma_f32_16x16x32_f16(
                false, ua.h, false, ub.h, (short)0, acc[j], false, false);
        }
        __syncthreads();                       // done reading cur before overwrite
    }
}

// C 16x16 f32 layout: lanes 0-15 hold N=lane, M=i per VGPR i; lanes 16-31 hold M=8+i.
#define EPILOGUE_ITER(BODY)                                     \
    {   const int lane = threadIdx.x & 31;                      \
        const int half = lane >> 4;                             \
        const int lm   = lane & 15;                             \
        _Pragma("unroll")                                       \
        for (int j = 0; j < 4; ++j) {                           \
            int gn = n0 + j * 16 + lm;                          \
            _Pragma("unroll")                                   \
            for (int i = 0; i < 8; ++i) {                       \
                int row = m0 + half * 8 + i;                    \
                float v = acc[j][i];                            \
                BODY                                            \
            }                                                   \
        }                                                       \
    }

#define GEMM_PROLOGUE(KDIM)                                     \
    __shared__ _Float16 sB[2][64 * BSTRIDE];                    \
    int m0 = blockIdx.x * 64 + (threadIdx.x >> 5) * 16;         \
    int n0 = blockIdx.y * 64;                                   \
    v8f acc[4];                                                 \
    _Pragma("unroll")                                           \
    for (int j = 0; j < 4; ++j)                                 \
        _Pragma("unroll")                                       \
        for (int e = 0; e < 8; ++e) acc[j][e] = 0.0f;           \
    wmma_strip(A, Bw, (KDIM), m0, n0, sB[0], sB[1], acc);

// GEMM1: zxbcdt = u16 @ W_in^T ; route cols -> z (d_out) | xBC_raw | dt_raw
__global__ void gemm1_kernel(const _Float16* __restrict__ A, const _Float16* __restrict__ Bw,
                             float* __restrict__ zout, float* __restrict__ xbc,
                             float* __restrict__ dtr) {
    GEMM_PROLOGUE(DMODEL)
    EPILOGUE_ITER({
        if (gn < DINNER)                zout[(size_t)row * DINNER + gn] = v;
        else if (gn < DINNER + CONVDIM) xbc[(size_t)row * CONVDIM + (gn - DINNER)] = v;
        else if (gn < DINPROJ)          dtr[(size_t)row * NHEADS + (gn - DINNER - CONVDIM)] = v;
    })
}

// GEMM2: m = y16 @ W_out^T -> fp32 (d_out m region)
__global__ void gemm2_kernel(const _Float16* __restrict__ A, const _Float16* __restrict__ Bw,
                             float* __restrict__ out) {
    GEMM_PROLOGUE(DINNER)
    EPILOGUE_ITER({ out[(size_t)row * DMODEL + gn] = v; })
}

// GEMM3: g = gelu_tanh(h16 @ W1^T + b1) -> f16
__global__ void gemm3_kernel(const _Float16* __restrict__ A, const _Float16* __restrict__ Bw,
                             const float* __restrict__ bias, _Float16* __restrict__ out) {
    GEMM_PROLOGUE(DMODEL)
    EPILOGUE_ITER({
        float t = v + bias[gn];
        float g = 0.5f * t * (1.0f + tanhf(0.7978845608f * (t + 0.044715f * t * t * t)));
        out[(size_t)row * DFF + gn] = (_Float16)g;
    })
}

// GEMM4: x = g16 @ W2^T + b2 + 2*m -> fp32 (d_out x region)
__global__ void gemm4_kernel(const _Float16* __restrict__ A, const _Float16* __restrict__ Bw,
                             const float* __restrict__ bias, const float* __restrict__ mprev,
                             float* __restrict__ out) {
    GEMM_PROLOGUE(DFF)
    EPILOGUE_ITER({
        out[(size_t)row * DMODEL + gn] =
            v + bias[gn] + 2.0f * mprev[(size_t)row * DMODEL + gn];
    })
}

// ===========================================================================
// dt = softplus(dt_raw + dt_bias) -> d_out dt region
// ===========================================================================
__global__ void dt_softplus_kernel(const float* __restrict__ raw, const float* __restrict__ bias,
                                   float* __restrict__ out) {
    int idx = blockIdx.x * blockDim.x + threadIdx.x;
    if (idx >= ROWS * NHEADS) return;
    float v = raw[idx] + bias[idx & (NHEADS - 1)];
    out[idx] = (v > 0.0f) ? v + log1pf(expf(-v)) : log1pf(expf(v));
}

// ===========================================================================
// Causal depthwise conv (K=4) + bias + silu + mask
// ===========================================================================
__global__ void conv_silu_kernel(const float* __restrict__ raw, const float* __restrict__ cw,
                                 const float* __restrict__ cb, const float* __restrict__ mask,
                                 float* __restrict__ out) {
    int idx = blockIdx.x * blockDim.x + threadIdx.x;
    if (idx >= ROWS * CONVDIM) return;
    int c  = idx % CONVDIM;
    int rl = idx / CONVDIM;
    int l  = rl % SEQ;
    int b  = rl / SEQ;
    float acc = cb[c];
#pragma unroll
    for (int i = 0; i < 4; ++i) {
        int ll = l - 3 + i;
        if (ll >= 0) acc += raw[((size_t)b * SEQ + ll) * CONVDIM + c] * cw[c * 4 + i];
    }
    float s = acc / (1.0f + expf(-acc));
    out[idx] = s * mask[rl];
}

// ===========================================================================
// SSM scans: block = (b,h), 64 threads (one per p), 64-state vector in regs.
// Forward writes y_f + D*x ; backward adds y_b in-place (deterministic).
// ===========================================================================
__global__ void scan_fwd_kernel(const float* __restrict__ xc, const float* __restrict__ dtp,
                                const float* __restrict__ A_log, const float* __restrict__ Dp,
                                float* __restrict__ y) {
    const int b = blockIdx.x >> 4, h = blockIdx.x & 15, p = threadIdx.x;
    const float Ah = -expf(A_log[h]);
    const float Dh = Dp[h];
    __shared__ float sB[64], sC[64];
    float hs[64];
#pragma unroll
    for (int n = 0; n < 64; ++n) hs[n] = 0.0f;
    const float* base = xc  + (size_t)b * SEQ * CONVDIM;
    const float* dtb  = dtp + (size_t)b * SEQ * NHEADS + h;
    float*       yb   = y   + (size_t)b * SEQ * DINNER + h * 64 + p;
    for (int t = 0; t < SEQ; ++t) {
        const float* row = base + (size_t)t * CONVDIM;
        sB[p] = row[DINNER + p];
        sC[p] = row[DINNER + DSTATE + p];
        __syncthreads();
        float dt  = dtb[(size_t)t * NHEADS];
        float xv  = row[h * 64 + p];
        float dA  = expf(dt * Ah);
        float dBx = dt * xv;
        float acc = 0.0f;
#pragma unroll
        for (int n = 0; n < 64; ++n) {
            hs[n] = fmaf(dA, hs[n], dBx * sB[n]);
            acc   = fmaf(hs[n], sC[n], acc);
        }
        yb[(size_t)t * DINNER] = fmaf(Dh, xv, acc);
        __syncthreads();
    }
}

__global__ void scan_bwd_kernel(const float* __restrict__ xc, const float* __restrict__ dtp,
                                const float* __restrict__ A_log, float* __restrict__ y) {
    const int b = blockIdx.x >> 4, h = blockIdx.x & 15, p = threadIdx.x;
    const float Ah = -expf(A_log[h]);
    __shared__ float sB[64], sC[64];
    float hs[64];
#pragma unroll
    for (int n = 0; n < 64; ++n) hs[n] = 0.0f;
    const float* base = xc  + (size_t)b * SEQ * CONVDIM;
    const float* dtb  = dtp + (size_t)b * SEQ * NHEADS + h;
    float*       yb   = y   + (size_t)b * SEQ * DINNER + h * 64 + p;
    for (int t = SEQ - 1; t >= 0; --t) {
        const float* row = base + (size_t)t * CONVDIM;
        sB[p] = row[DINNER + p];
        sC[p] = row[DINNER + DSTATE + p];
        __syncthreads();
        float dt  = dtb[(size_t)t * NHEADS];
        float xv  = row[h * 64 + p];
        float dA  = expf(dt * Ah);
        float dBx = dt * xv;
        float acc = 0.0f;
#pragma unroll
        for (int n = 0; n < 64; ++n) {
            hs[n] = fmaf(dA, hs[n], dBx * sB[n]);
            acc   = fmaf(hs[n], sC[n], acc);
        }
        yb[(size_t)t * DINNER] += acc;   // finalize: y_f + D*x + y_b
        __syncthreads();
    }
}

// ===========================================================================
// Gate + rmsnorm: y16 = rmsnorm(y * silu(z)) * norm_w  -> f16
// ===========================================================================
__global__ void gate_norm_kernel(const float* __restrict__ y, const float* __restrict__ z,
                                 const float* __restrict__ nw, _Float16* __restrict__ out) {
    int row = blockIdx.x, tid = threadIdx.x;
    const float* yr = y + (size_t)row * DINNER;
    const float* zr = z + (size_t)row * DINNER;
    float v[4], ss = 0.0f;
#pragma unroll
    for (int i = 0; i < 4; ++i) {
        int c = tid + i * 256;
        float zv = zr[c];
        float sz = zv / (1.0f + expf(-zv));
        float vv = yr[c] * sz;
        v[i] = vv; ss += vv * vv;
    }
    __shared__ float red[256];
    red[tid] = ss; __syncthreads();
    for (int st = 128; st > 0; st >>= 1) { if (tid < st) red[tid] += red[tid + st]; __syncthreads(); }
    float rs = rsqrtf(red[0] * (1.0f / (float)DINNER) + EPSF);
#pragma unroll
    for (int i = 0; i < 4; ++i) {
        int c = tid + i * 256;
        out[(size_t)row * DINNER + c] = (_Float16)(v[i] * rs * nw[c]);
    }
}

// ===========================================================================
// Launch
// ===========================================================================
extern "C" void kernel_launch(void* const* d_in, const int* in_sizes, int n_in,
                              void* d_out, int out_size, void* d_ws, size_t ws_size,
                              hipStream_t stream) {
    const float* x       = (const float*)d_in[0];
    const float* mask    = (const float*)d_in[1];
    const float* W_in    = (const float*)d_in[2];
    const float* conv_w  = (const float*)d_in[3];
    const float* conv_b  = (const float*)d_in[4];
    const float* dt_bias = (const float*)d_in[5];
    const float* A_log   = (const float*)d_in[6];
    const float* Dv      = (const float*)d_in[7];
    const float* norm_w  = (const float*)d_in[8];
    const float* W_out   = (const float*)d_in[9];
    const float* W1      = (const float*)d_in[10];
    const float* b1      = (const float*)d_in[11];
    const float* W2      = (const float*)d_in[12];
    const float* b2      = (const float*)d_in[13];

    char* ws = (char*)d_ws;
    _Float16* wtin  = (_Float16*)(ws + WS_WTIN);
    _Float16* wtout = (_Float16*)(ws + WS_WTOUT);
    _Float16* wt1   = (_Float16*)(ws + WS_WT1);
    _Float16* wt2   = (_Float16*)(ws + WS_WT2);
    _Float16* u16   = (_Float16*)(ws + WS_U16);     // also h16 later
    float*    xraw  = (float*)   (ws + WS_XRAW);
    _Float16* g16   = (_Float16*)(ws + WS_XRAW);    // reuse after conv consumed
    float*    xconv = (float*)   (ws + WS_XCONV);
    float*    dtraw = (float*)   (ws + WS_DTRAW);
    float*    ybuf  = (float*)   (ws + WS_Y);
    _Float16* y16   = (_Float16*)(ws + WS_Y16);

    float* xout  = (float*)d_out + X_OFF;
    float* mout  = (float*)d_out + M_OFF;
    float* dtout = (float*)d_out + DT_OFF;
    float* zout  = (float*)d_out + Z_OFF;

    // 1) weight transpose + f16 convert (Bw layout = [N][K])
    transpose_to_f16<<<(NPAD1 * DMODEL + 255) / 256, 256, 0, stream>>>(W_in,  wtin,  DMODEL, DINPROJ, NPAD1);
    transpose_to_f16<<<(DMODEL * DINNER + 255) / 256, 256, 0, stream>>>(W_out, wtout, DINNER, DMODEL, DMODEL);
    transpose_to_f16<<<(DFF * DMODEL + 255) / 256, 256, 0, stream>>>(W1,    wt1,   DMODEL, DFF,    DFF);
    transpose_to_f16<<<(DMODEL * DFF + 255) / 256, 256, 0, stream>>>(W2,    wt2,   DFF,    DMODEL, DMODEL);

    // 2) u = rmsnorm(x * mask) -> f16
    rmsnorm_f16_kernel<<<ROWS, 256, 0, stream>>>(x, mask, 1.0f, DMODEL, u16);

    // 3) zxbcdt = u @ W_in  (routes z -> d_out, xBC -> ws, dt_raw -> ws)
    gemm1_kernel<<<dim3(ROWS / 64, NPAD1 / 64), 128, 0, stream>>>(u16, wtin, zout, xraw, dtraw);

    // 4) dt = softplus(dt_raw + dt_bias) -> d_out dt region
    dt_softplus_kernel<<<(ROWS * NHEADS + 255) / 256, 256, 0, stream>>>(dtraw, dt_bias, dtout);

    // 5) xBC = silu(causal_conv(xBC) + conv_b) * mask
    conv_silu_kernel<<<(ROWS * CONVDIM + 255) / 256, 256, 0, stream>>>(xraw, conv_w, conv_b, mask, xconv);

    // 6/7) bidirectional SSM scan (sequential over L, parallel over b*h*p)
    scan_fwd_kernel<<<BATCH * NHEADS, 64, 0, stream>>>(xconv, dtout, A_log, Dv, ybuf);
    scan_bwd_kernel<<<BATCH * NHEADS, 64, 0, stream>>>(xconv, dtout, A_log, ybuf);

    // 8) y16 = rmsnorm(y * silu(z)) * norm_w
    gate_norm_kernel<<<ROWS, 256, 0, stream>>>(ybuf, zout, norm_w, y16);

    // 9) m = y16 @ W_out -> d_out m region
    gemm2_kernel<<<dim3(ROWS / 64, DMODEL / 64), 128, 0, stream>>>(y16, wtout, mout);

    // 10) h = rmsnorm(2*m) -> f16 (reuse u16)
    rmsnorm_f16_kernel<<<ROWS, 256, 0, stream>>>(mout, nullptr, 2.0f, DMODEL, u16);

    // 11) g = gelu(h @ W1 + b1) -> f16 (reuse xraw region)
    gemm3_kernel<<<dim3(ROWS / 64, DFF / 64), 128, 0, stream>>>(u16, wt1, b1, g16);

    // 12) x = 2*m + g @ W2 + b2 -> d_out x region
    gemm4_kernel<<<dim3(ROWS / 64, DMODEL / 64), 128, 0, stream>>>(g16, wt2, b2, mout, xout);
}
